// Attn_43980465111716
// MI455X (gfx1250) — matrix operand order repaired and checked by reading, not compiled, not run on hardware
//
#include <hip/hip_runtime.h>
#include <hip/hip_bf16.h>

// ---------------------------------------------------------------------------
// Causal multi-head attention (B=2, S=4096, D=512, H=8, HD=64) for gfx1250.
// bf16 WMMA (v_wmma_f32_16x16x32_bf16) with f32 accumulation everywhere.
// Pipeline: QKV GEMM -> flash attention -> output GEMM.  Workspace: 32 MB bf16.
// GEMMs use a 2x2 register tile per wave (4 WMMA per K-step, fragment reuse).
// V-tile staging uses GLOBAL_LOAD_ASYNC_TO_LDS_B128 (ASYNCcnt path).
// ---------------------------------------------------------------------------

#define S_LEN   4096
#define DMODEL  512
#define NH      8
#define HDIM    64
#define BATCH   2
#define MTOT    (BATCH * S_LEN)        // 8192
#define NQKV    (3 * DMODEL)           // 1536

// GEMM tiling: block 128x64, wave 32x32 (2x2 of 16x16 WMMA tiles), BK=32
#define BM 128
#define BN 64
#define BK 32
#define APITCH 40
#define BPITCH 72

// Attention tiling
#define QTILE 128
#define KVT   32
#define KTP   40     // sKT pitch (bf16 elems)
#define VP    72     // sV  pitch
#define PP    40     // sP  pitch

typedef __attribute__((ext_vector_type(16))) __bf16 v16bf;
typedef __attribute__((ext_vector_type(8)))  float  v8f;

union AF { v16bf v; unsigned int u[8]; };
union BF { v16bf v; unsigned int u[8]; };

__device__ __forceinline__ unsigned f2bfbits(float f) {
    union { float f; unsigned u; } a; a.f = f;
    return (a.u + 0x7FFFu + ((a.u >> 16) & 1u)) >> 16;        // RNE, low 16 bits
}

__device__ __forceinline__ __bf16 f2bf(float f) {
    union { unsigned short s; __bf16 b; } o;
    o.s = (unsigned short)f2bfbits(f);
    return o.b;
}

__device__ __forceinline__ v8f wmma_bf16(const AF& a, const BF& b, v8f c) {
    return __builtin_amdgcn_wmma_f32_16x16x32_bf16(
        /*neg_a*/false, a.v, /*neg_b*/false, b.v,
        /*c_mod*/(short)0, c, /*reuse_a*/false, /*reuse_b*/false);
}

// A-fragment K index for VGPR pair v (e = 2v), given lane half.
__device__ __forceinline__ int akb(int v, int hv) {
    return 2 * v + (v >= 4 ? 8 : 0) + 8 * hv;
}

// ---------------------------------------------------------------------------
// Kernel 1: qkv = x @ Wqkv^T + bqkv, scattered to bf16 Q/K/V in [B*H, S, 64].
// ---------------------------------------------------------------------------
__global__ __launch_bounds__(256) void qkv_gemm_kernel(
    const float* __restrict__ x,      // [8192, 512]
    const float* __restrict__ Wqkv,   // [1536, 512]
    const float* __restrict__ bqkv,   // [1536]
    __bf16* __restrict__ qbuf, __bf16* __restrict__ kbuf, __bf16* __restrict__ vbuf)
{
    __shared__ __bf16 sA[BM * APITCH];     // [m][k]
    __shared__ __bf16 sB[BK * BPITCH];     // [k][n] (transposed Wqkv tile)

    const int tid  = threadIdx.x;
    const int wave = tid >> 5, lane = tid & 31;
    const int m0 = blockIdx.y * BM;
    const int n0 = blockIdx.x * BN;
    const int mw = (wave >> 1) * 32;
    const int nw = (wave & 1) * 32;
    const int mA = lane & 15, hv = lane >> 4, nl = lane & 15;

    const int arow = tid >> 1;             // 0..127
    const int acol = (tid & 1) * 16;       // 0 or 16
    const int brow = tid >> 2;             // 0..63 (n)
    const int bcol = (tid & 3) * 8;        // 0..24 (k)

    const v8f vzero = {0.f, 0.f, 0.f, 0.f, 0.f, 0.f, 0.f, 0.f};
    v8f acc[2][2];
    #pragma unroll
    for (int mi = 0; mi < 2; ++mi)
        #pragma unroll
        for (int ni = 0; ni < 2; ++ni) acc[mi][ni] = vzero;

    for (int k0 = 0; k0 < DMODEL; k0 += BK) {
        // stage A (f32 -> packed bf16 pairs -> two b128 LDS stores)
        {
            const float* ap = x + (size_t)(m0 + arow) * DMODEL + k0 + acol;
            unsigned pk[8];
            #pragma unroll
            for (int i = 0; i < 8; ++i)
                pk[i] = f2bfbits(ap[2 * i]) | (f2bfbits(ap[2 * i + 1]) << 16);
            uint4* dst = (uint4*)&sA[arow * APITCH + acol];
            dst[0] = *(uint4*)&pk[0];
            dst[1] = *(uint4*)&pk[4];
        }
        // stage B transposed
        {
            const float* bp = Wqkv + (size_t)(n0 + brow) * DMODEL + k0 + bcol;
            #pragma unroll
            for (int j = 0; j < 8; ++j)
                sB[(bcol + j) * BPITCH + brow] = f2bf(bp[j]);
        }
        __syncthreads();

        AF a[2]; BF b[2];
        #pragma unroll
        for (int mi = 0; mi < 2; ++mi)
            #pragma unroll
            for (int v = 0; v < 8; ++v)
                a[mi].u[v] = *(const unsigned*)
                    &sA[(mw + 16 * mi + mA) * APITCH + akb(v, hv)];
        #pragma unroll
        for (int ni = 0; ni < 2; ++ni)
            #pragma unroll
            for (int v = 0; v < 8; ++v)
                b[ni].u[v] = (unsigned)*(const unsigned short*)&sB[akb(v, hv) * BPITCH + nw + 16 * ni + nl]
                    | ((unsigned)*(const unsigned short*)&sB[(akb(v, hv) + 1) * BPITCH + nw + 16 * ni + nl] << 16);

        #pragma unroll
        for (int mi = 0; mi < 2; ++mi)
            #pragma unroll
            for (int ni = 0; ni < 2; ++ni)
                acc[mi][ni] = wmma_bf16(a[mi], b[ni], acc[mi][ni]);
        __syncthreads();
    }

    #pragma unroll
    for (int mi = 0; mi < 2; ++mi)
        #pragma unroll
        for (int ni = 0; ni < 2; ++ni)
            #pragma unroll
            for (int r = 0; r < 8; ++r) {
                int m = m0 + mw + 16 * mi + r + 8 * hv;
                int n = n0 + nw + 16 * ni + nl;
                float val = acc[mi][ni][r] + bqkv[n];
                int b_  = m >> 12;                 // / 4096
                int s_  = m & (S_LEN - 1);
                int sec = n >> 9;                  // 0=Q 1=K 2=V
                int h_  = (n & 511) >> 6;
                int hd_ = n & 63;
                __bf16* dst = (sec == 0) ? qbuf : ((sec == 1) ? kbuf : vbuf);
                dst[((size_t)(b_ * NH + h_) * S_LEN + s_) * HDIM + hd_] = f2bf(val);
            }
}

// ---------------------------------------------------------------------------
// Kernel 2: causal flash attention.  grid = (S/128, B*H), 256 thr (8 waves).
// Each wave owns 16 query rows; streams kv in tiles of 32 with online softmax.
// ---------------------------------------------------------------------------
__global__ __launch_bounds__(256) void attn_kernel(
    const __bf16* __restrict__ qbuf, const __bf16* __restrict__ kbuf,
    const __bf16* __restrict__ vbuf, __bf16* __restrict__ ybuf)
{
    __shared__ __bf16 sKT[HDIM * KTP];        // [d][kv] transposed K tile
    __shared__ __bf16 sV [KVT * VP];          // [kv][d]
    __shared__ __bf16 sP [8 * 16 * PP];       // per-wave P (16 x 32)

    const int tid  = threadIdx.x;
    const int wave = tid >> 5, lane = tid & 31;
    const int bh   = blockIdx.y;
    const int q0b  = blockIdx.x * QTILE;
    const int q0w  = q0b + wave * 16;
    const int mA = lane & 15, hv = lane >> 4, nl = lane & 15;

    const __bf16* qp = qbuf + (size_t)bh * S_LEN * HDIM;
    const __bf16* kp = kbuf + (size_t)bh * S_LEN * HDIM;
    const __bf16* vp = vbuf + (size_t)bh * S_LEN * HDIM;

    // Q fragments: 16 rows x 64 d, as two K=32 A-fragments (kept in VGPRs).
    AF qf[2];
    #pragma unroll
    for (int c = 0; c < 2; ++c)
        #pragma unroll
        for (int v = 0; v < 8; ++v)
            qf[c].u[v] = *(const unsigned*)(qp + (size_t)(q0w + mA) * HDIM
                                               + 32 * c + akb(v, hv));

    const v8f vzero = {0.f, 0.f, 0.f, 0.f, 0.f, 0.f, 0.f, 0.f};
    v8f vy[4];
    #pragma unroll
    for (int nc = 0; nc < 4; ++nc) vy[nc] = vzero;
    float mrow[8], lrow[8];
    #pragma unroll
    for (int r = 0; r < 8; ++r) { mrow[r] = -1e30f; lrow[r] = 0.f; }

    const int skv = tid >> 3;            // 0..31
    const int sd  = (tid & 7) * 8;       // 0..56

    const int kv_end = q0b + QTILE;      // causal block limit (<= S_LEN)
    for (int kv0 = 0; kv0 < kv_end; kv0 += KVT) {
        // ---- stage K (transposed, regular stores) ----
        {
            const __bf16* krow = kp + (size_t)(kv0 + skv) * HDIM + sd;
            __builtin_prefetch(krow + KVT * HDIM, 0, 0);
            #pragma unroll
            for (int j = 0; j < 8; ++j)
                sKT[(sd + j) * KTP + skv] = krow[j];
        }
        // ---- stage V via async global->LDS DMA (ASYNCcnt path) ----
        {
            const __bf16* vrow = vp + (size_t)(kv0 + skv) * HDIM + sd;
            unsigned ldsa = (unsigned)(unsigned long long)&sV[skv * VP + sd];
            asm volatile("global_load_async_to_lds_b128 %0, %1, off"
                         :: "v"(ldsa), "v"((unsigned long long)vrow)
                         : "memory");
        }
        asm volatile("s_wait_asynccnt 0x0" ::: "memory");
        __syncthreads();

        const bool active = (kv0 <= q0w + 15);   // wave-uniform causal skip
        if (active) {
            // ---- S = (Q @ K^T) * scale, over two 16-wide kv sub-tiles ----
            v8f st[2];
            #pragma unroll
            for (int t = 0; t < 2; ++t) {
                v8f sc = vzero;
                #pragma unroll
                for (int c = 0; c < 2; ++c) {
                    BF b;
                    #pragma unroll
                    for (int v = 0; v < 8; ++v)
                        b.u[v] = (unsigned)*(const unsigned short*)&sKT[(32 * c + akb(v, hv)) * KTP + 16 * t + nl]
                            | ((unsigned)*(const unsigned short*)&sKT[(32 * c + akb(v, hv) + 1) * KTP + 16 * t + nl] << 16);
                    sc = wmma_bf16(qf[c], b, sc);
                }
                st[t] = sc;
            }
            // ---- online softmax update + write P to per-wave LDS ----
            #pragma unroll
            for (int r = 0; r < 8; ++r) {
                const int q = q0w + r + 8 * hv;
                float a0 = st[0][r] * 0.125f;          // 1/sqrt(64)
                float a1 = st[1][r] * 0.125f;
                if (kv0 + nl      > q) a0 = -1e30f;
                if (kv0 + 16 + nl > q) a1 = -1e30f;
                float mx = fmaxf(a0, a1);
                #pragma unroll
                for (int off = 1; off < 16; off <<= 1)
                    mx = fmaxf(mx, __shfl_xor(mx, off, 32));
                const float mnew  = fmaxf(mrow[r], mx);
                const float alpha = __expf(mrow[r] - mnew);
                const float p0 = __expf(a0 - mnew);
                const float p1 = __expf(a1 - mnew);
                float ps = p0 + p1;
                #pragma unroll
                for (int off = 1; off < 16; off <<= 1)
                    ps += __shfl_xor(ps, off, 32);
                lrow[r] = lrow[r] * alpha + ps;
                mrow[r] = mnew;
                #pragma unroll
                for (int nc = 0; nc < 4; ++nc) vy[nc][r] *= alpha;
                sP[(wave * 16 + r + 8 * hv) * PP + nl]      = f2bf(p0);
                sP[(wave * 16 + r + 8 * hv) * PP + 16 + nl] = f2bf(p1);
            }
        }
        __syncthreads();   // P visible (block-uniform barrier)

        if (active) {
            // ---- y += P @ V ----
            AF pf;
            #pragma unroll
            for (int v = 0; v < 8; ++v)
                pf.u[v] = *(const unsigned*)&sP[(wave * 16 + mA) * PP + akb(v, hv)];
            #pragma unroll
            for (int nc = 0; nc < 4; ++nc) {
                BF b;
                #pragma unroll
                for (int v = 0; v < 8; ++v)
                    b.u[v] = (unsigned)*(const unsigned short*)&sV[akb(v, hv) * VP + 16 * nc + nl]
                        | ((unsigned)*(const unsigned short*)&sV[(akb(v, hv) + 1) * VP + 16 * nc + nl] << 16);
                vy[nc] = wmma_bf16(pf, b, vy[nc]);
            }
        }
        __syncthreads();   // before overwriting sKT/sV/sP
    }

    // ---- normalize and store y in [B, S, D] (bf16) ----
    const int b_ = bh >> 3;          // bh / NH
    const int h_ = bh & (NH - 1);
    #pragma unroll
    for (int r = 0; r < 8; ++r) {
        const int q = q0w + r + 8 * hv;
        const float inv = 1.0f / lrow[r];
        #pragma unroll
        for (int nc = 0; nc < 4; ++nc) {
            float val = vy[nc][r] * inv;
            ybuf[((size_t)(b_ * S_LEN + q)) * DMODEL + h_ * HDIM + nc * 16 + nl]
                = f2bf(val);
        }
    }
}

// ---------------------------------------------------------------------------
// Kernel 3: out = y @ Wo^T + bo  (y bf16 from workspace, out fp32).
// ---------------------------------------------------------------------------
__global__ __launch_bounds__(256) void out_gemm_kernel(
    const __bf16* __restrict__ y,     // [8192, 512] bf16
    const float*  __restrict__ Wo,    // [512, 512]
    const float*  __restrict__ bo,    // [512]
    float* __restrict__ out)          // [8192, 512]
{
    __shared__ __bf16 sA[BM * APITCH];
    __shared__ __bf16 sB[BK * BPITCH];

    const int tid  = threadIdx.x;
    const int wave = tid >> 5, lane = tid & 31;
    const int m0 = blockIdx.y * BM;
    const int n0 = blockIdx.x * BN;
    const int mw = (wave >> 1) * 32;
    const int nw = (wave & 1) * 32;
    const int mA = lane & 15, hv = lane >> 4, nl = lane & 15;

    const int arow = tid >> 1;             // 0..127
    const int acol = (tid & 1) * 16;       // 0 or 16
    const int brow = tid >> 2;             // 0..63 (n)
    const int bcol = (tid & 3) * 8;        // 0..24 (k)

    const v8f vzero = {0.f, 0.f, 0.f, 0.f, 0.f, 0.f, 0.f, 0.f};
    v8f acc[2][2];
    #pragma unroll
    for (int mi = 0; mi < 2; ++mi)
        #pragma unroll
        for (int ni = 0; ni < 2; ++ni) acc[mi][ni] = vzero;

    for (int k0 = 0; k0 < DMODEL; k0 += BK) {
        // stage A (already bf16): two b128 copies
        {
            const __bf16* ap = y + (size_t)(m0 + arow) * DMODEL + k0 + acol;
            uint4* dst = (uint4*)&sA[arow * APITCH + acol];
            dst[0] = ((const uint4*)ap)[0];
            dst[1] = ((const uint4*)ap)[1];
        }
        // stage B transposed (f32 -> bf16)
        {
            const float* bp = Wo + (size_t)(n0 + brow) * DMODEL + k0 + bcol;
            #pragma unroll
            for (int j = 0; j < 8; ++j)
                sB[(bcol + j) * BPITCH + brow] = f2bf(bp[j]);
        }
        __syncthreads();

        AF a[2]; BF b[2];
        #pragma unroll
        for (int mi = 0; mi < 2; ++mi)
            #pragma unroll
            for (int v = 0; v < 8; ++v)
                a[mi].u[v] = *(const unsigned*)
                    &sA[(mw + 16 * mi + mA) * APITCH + akb(v, hv)];
        #pragma unroll
        for (int ni = 0; ni < 2; ++ni)
            #pragma unroll
            for (int v = 0; v < 8; ++v)
                b[ni].u[v] = (unsigned)*(const unsigned short*)&sB[akb(v, hv) * BPITCH + nw + 16 * ni + nl]
                    | ((unsigned)*(const unsigned short*)&sB[(akb(v, hv) + 1) * BPITCH + nw + 16 * ni + nl] << 16);

        #pragma unroll
        for (int mi = 0; mi < 2; ++mi)
            #pragma unroll
            for (int ni = 0; ni < 2; ++ni)
                acc[mi][ni] = wmma_bf16(a[mi], b[ni], acc[mi][ni]);
        __syncthreads();
    }

    #pragma unroll
    for (int mi = 0; mi < 2; ++mi)
        #pragma unroll
        for (int ni = 0; ni < 2; ++ni)
            #pragma unroll
            for (int r = 0; r < 8; ++r) {
                int m = m0 + mw + 16 * mi + r + 8 * hv;
                int n = n0 + nw + 16 * ni + nl;
                out[(size_t)m * DMODEL + n] = acc[mi][ni][r] + bo[n];
            }
}

// ---------------------------------------------------------------------------
// Launcher.  Workspace layout (bf16): Q | K | V | y  (4 x 4,194,304 = 32 MB).
// ---------------------------------------------------------------------------
extern "C" void kernel_launch(void* const* d_in, const int* in_sizes, int n_in,
                              void* d_out, int out_size, void* d_ws, size_t ws_size,
                              hipStream_t stream) {
    const float* x    = (const float*)d_in[0];
    const float* Wqkv = (const float*)d_in[1];
    const float* bqkv = (const float*)d_in[2];
    const float* Wo   = (const float*)d_in[3];
    const float* bo   = (const float*)d_in[4];
    float* out = (float*)d_out;

    const size_t NBSD = (size_t)BATCH * S_LEN * DMODEL;   // 4,194,304
    __bf16* ws = (__bf16*)d_ws;
    __bf16* qb = ws;
    __bf16* kb = ws + NBSD;
    __bf16* vb = ws + 2 * NBSD;
    __bf16* yb = ws + 3 * NBSD;

    dim3 g1(NQKV / BN, MTOT / BM);            // (24, 64)
    qkv_gemm_kernel<<<g1, 256, 0, stream>>>(x, Wqkv, bqkv, qb, kb, vb);

    dim3 g2(S_LEN / QTILE, BATCH * NH);       // (32, 16)
    attn_kernel<<<g2, 256, 0, stream>>>(qb, kb, vb, yb);

    dim3 g3(DMODEL / BN, MTOT / BM);          // (8, 64)
    out_gemm_kernel<<<g3, 256, 0, stream>>>(yb, Wo, bo, out);
}
